// NN_transition_WFA_75711683494391
// MI455X (gfx1250) — matrix operand order repaired and checked by reading, not compiled, not run on hardware
//
#include <hip/hip_runtime.h>
#include <hip/hip_bf16.h>

// ---------------------------------------------------------------------------
// MI455X (gfx1250) fused implementation of the NN-transition-WFA reference.
//
// ~48 GFLOP total, ~22 MB resident weights (L2 = 192 MB) -> compute-bound ->
// all GEMMs on v_wmma_f32_16x16x32_f16 (fp16 in, fp32 accumulate).
//
// B operands are pre-swizzled once into WMMA-fragment-native 32x16 tiles:
//     tile[lane*16 + e] = W[K = kt*32 + (lane>=16?16:0) + e][jt*16 + lane%16]
// so LDS staging is contiguous and every B-fragment load is one aligned v16h
// LDS read (2x ds_load_b128). LDS staging itself uses the CDNA5 async
// Global->LDS DMA (global_load_async_to_lds_b128, ASYNCcnt-tracked),
// software-pipelined ahead of the WMMA consumption.
//
// wave32: blockDim=128 => 4 waves, one 16-row M-tile per wave, 64 rows/block.
// ---------------------------------------------------------------------------

typedef _Float16 f16_t;
typedef __attribute__((ext_vector_type(16))) _Float16 v16h;
typedef __attribute__((ext_vector_type(8)))  float    v8f;
typedef __attribute__((ext_vector_type(4)))  int      i4vec;

#if defined(__HIP_DEVICE_COMPILE__) && \
    __has_builtin(__builtin_amdgcn_global_load_async_to_lds_b128)
#define USE_ASYNC_LDS 1
#else
#define USE_ASYNC_LDS 0
#endif

#if USE_ASYNC_LDS
typedef __attribute__((address_space(1))) i4vec* gb128_t;  // global int4*
typedef __attribute__((address_space(3))) i4vec* lb128_t;  // LDS int4*

// Recursive template so every async offset is an integer constant expression.
template <int K, int N>
struct AsyncB128 {
  static __device__ __forceinline__ void run(gb128_t gp, lb128_t lp) {
    __builtin_amdgcn_global_load_async_to_lds_b128(gp, lp, K * 2048, 0);
    AsyncB128<K + 1, N>::run(gp, lp);
  }
};
template <int N>
struct AsyncB128<N, N> {
  static __device__ __forceinline__ void run(gb128_t, lb128_t) {}
};
#endif

// Issue a BYTES-sized contiguous Global->LDS copy with 128 threads.
// Async path: each thread issues BYTES/2048 async b128 DMAs (no VGPR traffic).
template <int BYTES>
__device__ inline void stage_issue(const f16_t* g, f16_t* l) {
#if USE_ASYNC_LDS
  gb128_t gp = (gb128_t)((const char*)g + threadIdx.x * 16);
  lb128_t lp = (lb128_t)((char*)l + threadIdx.x * 16);
  AsyncB128<0, BYTES / 2048>::run(gp, lp);
#else
  const uint4* gs = (const uint4*)g;
  uint4* ls = (uint4*)l;
  for (int t = threadIdx.x; t < BYTES / 16; t += 128) ls[t] = gs[t];
#endif
}

__device__ inline void stage_wait() {
#if USE_ASYNC_LDS
#if __has_builtin(__builtin_amdgcn_s_wait_asynccnt)
  __builtin_amdgcn_s_wait_asynccnt(0);
#else
  asm volatile("s_wait_asynccnt 0x0" ::: "memory");
#endif
#endif
}

// ---- WMMA fragment helpers (contiguous k = 16*(lane/16) + e, same for A and B) ------------------

// A-matrix 16x32 f16 from a row-major LDS tile (ld = row stride in halfs).
__device__ inline v16h ld_a16(const f16_t* p, int ld) {
  const int lane = threadIdx.x & 31;
  const f16_t* r = p + (lane & 15) * ld + ((lane & 16) ? 16 : 0);
  v16h a;
#pragma unroll
  for (int e = 0; e < 8; ++e)  a[e] = r[e];
#pragma unroll
  for (int e = 8; e < 16; ++e) a[e] = r[e];
  return a;
}

// Same but source tile is f32 (converted on the fly).
__device__ inline v16h ld_a16_f32(const float* p, int ld) {
  const int lane = threadIdx.x & 31;
  const float* r = p + (lane & 15) * ld + ((lane & 16) ? 16 : 0);
  v16h a;
#pragma unroll
  for (int e = 0; e < 8; ++e)  a[e] = (f16_t)r[e];
#pragma unroll
  for (int e = 8; e < 16; ++e) a[e] = (f16_t)r[e];
  return a;
}

// B-matrix 32x16 f16 from a fragment-native LDS tile -> 2x ds_load_b128.
__device__ inline v16h ld_b16v(const f16_t* tile) {
  const int lane = threadIdx.x & 31;
  return *reinterpret_cast<const v16h*>(tile + lane * 16);
}

__device__ inline v8f wmma16(v16h a, v16h b, v8f c) {
  return __builtin_amdgcn_wmma_f32_16x16x32_f16(false, a, false, b,
                                                (short)0, c, false, false);
}

// logsumexp over the 16-lane group (mixture comps live across D-frag lanes).
__device__ inline float lse16(float v) {
  float m = v;
#pragma unroll
  for (int off = 1; off < 16; off <<= 1) m = fmaxf(m, __shfl_xor(m, off, 16));
  float s = __expf(v - m);
#pragma unroll
  for (int off = 1; off < 16; off <<= 1) s += __shfl_xor(s, off, 16);
  return m + __logf(s);
}

// ---------------------------------------------------------------------------

struct StepParams {
  const float* X;          // [4096][8][9]
  float* h;                // [4096][64] state (read + written)
  const f16_t* core;       // this step's core, swizzled tiles, 8192 halfs / i
  const f16_t *e1, *e2;    // e1 linear [8][128]; e2 swizzled (K=128,J=128)
  const f16_t* nw;         // swizzled (K=64,J=256)
  const f16_t *mw, *sw, *aw;  // swizzled (K=256,J=128), jt-major
  const float *b1, *b2, *nb, *mb, *sb, *ab;
  const float *stats;      // [mean 64 | var 64]
  const float *gamma, *beta;
  float* out;              // [4096]
  int colEnc, colPhi;
};

// MODE 0: encode + bilinear step + phi          (normal timestep)
// MODE 1: encode + bilinear step only           (pre-BatchNorm, i == l)
// MODE 2: BatchNorm apply + phi                 (post-stats,    i == l)
template <int MODE>
__global__ __launch_bounds__(128, 1)
void fused_step(StepParams p) {
  extern __shared__ unsigned char smraw[];
  float* sXe   = reinterpret_cast<float*>(smraw);   // [64][8]
  float* sXp   = sXe + 512;                         // [64][8]
  float* sH    = sXp + 512;                         // [64][72] f32 state
  f16_t* sEncH = reinterpret_cast<f16_t*>(sH + 64 * 72);  // [64][136]
  f16_t* sEnc  = sEncH + 64 * 136;                  // [64][136]
  f16_t* sG    = sEnc + 64 * 136;                   // [64][264]
  f16_t* sStage = sG + 64 * 264;                    // 16384 halfs (32 KB)

  const int tid  = threadIdx.x;
  const int lane = tid & 31;
  const int wave = tid >> 5;
  const int row0 = wave * 16;          // this wave's 16-row M tile
  const int n0   = blockIdx.x * 64;    // block's batch rows
  const v8f vzero = {};

  // ---- async weight staging issued up front (overlaps the h/X loads) ------
  if (MODE != 2) {
    stage_issue<2048>(p.e1, sG);        // enc1 weights -> idle sG region
    stage_issue<32768>(p.e2, sStage);   // swizzled enc2 weights
  }

  // ---- cooperative loads ---------------------------------------------------
  if (MODE != 2) {
    for (int t = tid; t < 512; t += 128) {
      int r = t >> 3, xd = t & 7;
      sXe[t] = p.X[(size_t)(n0 + r) * 72 + xd * 9 + p.colEnc];
    }
  }
  if (MODE != 1) {
    for (int t = tid; t < 512; t += 128) {
      int r = t >> 3, xd = t & 7;
      sXp[t] = p.X[(size_t)(n0 + r) * 72 + xd * 9 + p.colPhi];
    }
  }
  for (int t = tid; t < 64 * 64; t += 128) {
    int r = t >> 6, c = t & 63;
    float v = p.h[(size_t)(n0 + r) * 64 + c];
    if (MODE == 2) {  // BatchNorm1d (training-mode stats from ws)
      float mean = p.stats[c], var = p.stats[64 + c];
      v = (v - mean) * rsqrtf(var + 1e-5f) * p.gamma[c] + p.beta[c];
      p.h[(size_t)(n0 + r) * 64 + c] = v;   // persist normalized state
    }
    sH[r * 72 + c] = v;
  }
  stage_wait();
  __syncthreads();

  if (MODE != 2) {
    // ---- enc1: [64x8]@[8x128] + relu (K=8 -> plain VALU, weights in LDS) ---
    for (int t = tid; t < 64 * 128; t += 128) {
      int r = t >> 7, j = t & 127;
      float s = p.b1[j];
#pragma unroll
      for (int k = 0; k < 8; ++k) s += sXe[r * 8 + k] * (float)sG[k * 128 + j];
      sEncH[r * 136 + j] = (f16_t)fmaxf(s, 0.f);
    }
    __syncthreads();

    // ---- enc2: [64x128]@[128x128] via WMMA ---------------------------------
    {
      v16h a4[4];
#pragma unroll
      for (int kt = 0; kt < 4; ++kt)
        a4[kt] = ld_a16(sEncH + row0 * 136 + kt * 32, 136);
#pragma unroll
      for (int jt = 0; jt < 8; ++jt) {
        v8f c = vzero;
#pragma unroll
        for (int kt = 0; kt < 4; ++kt)
          c = wmma16(a4[kt], ld_b16v(sStage + (jt * 4 + kt) * 512), c);
        int col = jt * 16 + (lane & 15);
        float bias = p.b2[col];
#pragma unroll
        for (int v = 0; v < 8; ++v) {
          int r = v + ((lane & 16) ? 8 : 0);
          sEnc[(row0 + r) * 136 + col] = (f16_t)(c[v] + bias);
        }
      }
    }
    __syncthreads();

    // ---- bilinear: h_new[n,j] = sum_{i,d} h[n,i] enc[n,d] core[i,d,j] ------
    // Software-pipelined async DMA: wait chunk i, barrier, issue chunk i+1
    // into the other buffer, then compute chunk i (DMA overlaps 16 WMMAs).
    v16h e4[4];
#pragma unroll
    for (int kt = 0; kt < 4; ++kt)
      e4[kt] = ld_a16(sEnc + row0 * 136 + kt * 32, 136);
    v8f hc[4];
#pragma unroll
    for (int jt = 0; jt < 4; ++jt) hc[jt] = vzero;

    stage_issue<16384>(p.core, sStage);           // chunk 0 -> buf 0
    for (int i = 0; i < 64; ++i) {
      f16_t* buf = sStage + (i & 1) * 8192;
      stage_wait();
      __syncthreads();
      if (i < 63)
        stage_issue<16384>(p.core + (size_t)(i + 1) * 8192,
                           sStage + ((i + 1) & 1) * 8192);
      f16_t hv = (f16_t)sH[(row0 + (lane & 15)) * 72 + i];
#pragma unroll
      for (int kt = 0; kt < 4; ++kt) {
        v16h a = e4[kt] * hv;
#pragma unroll
        for (int jt = 0; jt < 4; ++jt)
          hc[jt] = wmma16(a, ld_b16v(buf + (jt * 4 + kt) * 512), hc[jt]);
      }
    }
    __syncthreads();
    if (MODE == 0) stage_issue<32768>(p.nw, sStage);  // overlap with h store
    // write h_new (own rows only) to LDS + global state
#pragma unroll
    for (int jt = 0; jt < 4; ++jt) {
      int col = jt * 16 + (lane & 15);
#pragma unroll
      for (int v = 0; v < 8; ++v) {
        int r = v + ((lane & 16) ? 8 : 0);
        sH[(row0 + r) * 72 + col] = hc[jt][v];
        p.h[(size_t)(n0 + row0 + r) * 64 + col] = hc[jt][v];
      }
    }
    if (MODE == 1) return;   // compile-time uniform: all threads exit together
  }

  // ---- phi: g = relu(h @ nade_w + nb); MDN log-likelihood ------------------
  if (MODE == 2) stage_issue<32768>(p.nw, sStage);
  stage_wait();
  __syncthreads();
  {
    v16h ha0 = ld_a16_f32(sH + row0 * 72, 72);
    v16h ha1 = ld_a16_f32(sH + row0 * 72 + 32, 72);
#pragma unroll
    for (int jt = 0; jt < 16; ++jt) {
      v8f c = vzero;
      c = wmma16(ha0, ld_b16v(sStage + (jt * 2 + 0) * 512), c);
      c = wmma16(ha1, ld_b16v(sStage + (jt * 2 + 1) * 512), c);
      int col = jt * 16 + (lane & 15);
      float bias = p.nb[col];
#pragma unroll
      for (int v = 0; v < 8; ++v) {
        int r = v + ((lane & 16) ? 8 : 0);
        sG[(row0 + r) * 264 + col] = (f16_t)fmaxf(c[v] + bias, 0.f);
      }
    }
  }
  __syncthreads();

  v16h g8[8];
#pragma unroll
  for (int kt = 0; kt < 8; ++kt)
    g8[kt] = ld_a16(sG + row0 * 264 + kt * 32, 264);

  float racc[8] = {};
  for (int xd = 0; xd < 8; ++xd) {
    if (xd) __syncthreads();   // previous slice fully consumed
    // jt-major swizzle => per-xd slice of each head is contiguous 8 KB
    stage_issue<8192>(p.mw + (size_t)xd * 4096, sStage);
    stage_issue<8192>(p.sw + (size_t)xd * 4096, sStage + 4096);
    stage_issue<8192>(p.aw + (size_t)xd * 4096, sStage + 8192);
    stage_wait();
    __syncthreads();
    v8f cm = vzero, cs = vzero, ca = vzero;
#pragma unroll
    for (int kt = 0; kt < 8; ++kt) {
      cm = wmma16(g8[kt], ld_b16v(sStage +        kt * 512), cm);
      cs = wmma16(g8[kt], ld_b16v(sStage + 4096 + kt * 512), cs);
      ca = wmma16(g8[kt], ld_b16v(sStage + 8192 + kt * 512), ca);
    }
    int col = lane & 15;
    int cg  = xd * 16 + col;
    float mb = p.mb[cg], sb = p.sb[cg], ab = p.ab[cg];
#pragma unroll
    for (int v = 0; v < 8; ++v) {
      int r = v + ((lane & 16) ? 8 : 0);
      float mu = cm[v] + mb;
      float ls = cs[v] + sb;
      float al = ca[v] + ab;
      float la = al - lse16(al);                 // log_softmax over K comps
      float px = sXp[(row0 + r) * 8 + xd];
      float z  = (px - mu) * __expf(-ls);
      float comp = la - 0.5f * z * z - ls - 0.91893853320467274f;
      racc[v] += lse16(comp);                    // logsumexp over K comps
    }
  }
  if ((lane & 15) == 0) {
#pragma unroll
    for (int v = 0; v < 8; ++v) {
      int r = v + ((lane & 16) ? 8 : 0);
      p.out[n0 + row0 + r] += racc[v];
    }
  }
}

// ---------------------------------------------------------------------------

__global__ void cvt_f32_f16(const float* s, f16_t* d, int n) {
  for (int i = blockIdx.x * blockDim.x + threadIdx.x; i < n;
       i += gridDim.x * blockDim.x)
    d[i] = (f16_t)s[i];
}

// Convert + swizzle row-major [nmat][Kdim][Jdim] f32 weights into
// fragment-native 32x16 tiles, jt-major:
//   dst[m*K*J + (jt*ktiles + kt)*512 + lane*16 + e]
//     = src[m][kt*32 + (lane>=16 ? 16:0) + e][jt*16 + lane%16]
__global__ void swz_b16(const float* src, f16_t* dst, int Kdim, int Jdim,
                        int nmat) {
  const int ktiles = Kdim >> 5;
  const int per = Kdim * Jdim;
  const int total = per * nmat;
  for (int idx = blockIdx.x * blockDim.x + threadIdx.x; idx < total;
       idx += gridDim.x * blockDim.x) {
    int m = idx / per, rem = idx % per;
    int tile = rem >> 9;           // 512 halfs per tile
    int inner = rem & 511;
    int ln = inner >> 4, e = inner & 15;
    int jt = tile / ktiles, kt = tile % ktiles;
    int K = kt * 32 + ((ln & 16) ? 16 : 0) + e;
    int col = jt * 16 + (ln & 15);
    dst[idx] = (f16_t)src[(size_t)m * per + (size_t)K * Jdim + col];
  }
}

__global__ void init_state(const float* initw, float* h, float* out, int nrows) {
  int total = nrows * 64;
  for (int i = blockIdx.x * blockDim.x + threadIdx.x; i < total;
       i += gridDim.x * blockDim.x) {
    h[i] = initw[i & 63];
    if (i < nrows) out[i] = 0.f;
  }
}

// BatchNorm stats: 32-block partial sums + deterministic finalize.
__global__ void bn_partial(const float* h, float* part, int nrows) {
  __shared__ float acc[4][128];
  int c = threadIdx.x & 63, rg = threadIdx.x >> 6;   // 4 row-groups
  int chunk = nrows / gridDim.x;
  int rend = blockIdx.x * chunk + chunk;
  float s = 0.f, s2 = 0.f;
  for (int r = blockIdx.x * chunk + rg; r < rend; r += 4) {
    float v = h[(size_t)r * 64 + c];
    s += v; s2 += v * v;
  }
  acc[rg][c] = s; acc[rg][64 + c] = s2;
  __syncthreads();
  if (rg == 0) {
    part[blockIdx.x * 128 + c] =
        acc[0][c] + acc[1][c] + acc[2][c] + acc[3][c];
    part[blockIdx.x * 128 + 64 + c] =
        acc[0][64 + c] + acc[1][64 + c] + acc[2][64 + c] + acc[3][64 + c];
  }
}

__global__ void bn_final(const float* part, float* stats, int nrows,
                         int nparts) {
  int c = threadIdx.x;
  if (c >= 64) return;
  float s = 0.f, s2 = 0.f;
  for (int b = 0; b < nparts; ++b) {
    s += part[b * 128 + c];
    s2 += part[b * 128 + 64 + c];
  }
  float inv = 1.f / (float)nrows;
  float m = s * inv;
  stats[c] = m;
  stats[64 + c] = s2 * inv - m * m;
}

// ---------------------------------------------------------------------------

extern "C" void kernel_launch(void* const* d_in, const int* in_sizes, int n_in,
                              void* d_out, int out_size, void* d_ws, size_t ws_size,
                              hipStream_t stream) {
  const float* X      = (const float*)d_in[0];
  const float* initw  = (const float*)d_in[1];
  const float* cores  = (const float*)d_in[2];   // [8][64][128][64]
  const float* Amat   = (const float*)d_in[3];   // [64][128][64]
  const float* bng    = (const float*)d_in[4];
  const float* bnb    = (const float*)d_in[5];
  const float* e1w    = (const float*)d_in[6];
  const float* b1     = (const float*)d_in[7];
  const float* e2w    = (const float*)d_in[8];
  const float* b2     = (const float*)d_in[9];
  const float* nadew  = (const float*)d_in[10];
  const float* nb     = (const float*)d_in[11];
  const float* muw    = (const float*)d_in[12];
  const float* mb     = (const float*)d_in[13];
  const float* sigw   = (const float*)d_in[14];
  const float* sb     = (const float*)d_in[15];
  const float* alphaw = (const float*)d_in[16];
  const float* ab     = (const float*)d_in[17];
  float* out = (float*)d_out;

  const int N = in_sizes[0] / 72;          // 4096
  const int nblocks = N / 64;
  const int NPART = 32;

  // workspace: h state (f32) | stats (128) + pad | bn partials | f16 pool
  float* hws   = (float*)d_ws;                       // N*64 floats
  float* stats = hws + (size_t)N * 64;               // 128 floats (+pad)
  float* part  = stats + 256;                        // 32*128 floats
  f16_t* hb     = (f16_t*)((char*)d_ws +
                           ((size_t)N * 64 + 256 + 4096) * sizeof(float));
  f16_t* coresH = hb;                                // 8 * 524288 (swizzled)
  f16_t* AH     = hb + 8ull * 524288;                // 524288     (swizzled)
  f16_t* e1h    = AH + 524288;                       // 1024       (linear)
  f16_t* e2h    = e1h + 1024;                        // 16384      (swizzled)
  f16_t* nwh    = e2h + 16384;                       // 16384      (swizzled)
  f16_t* mwh    = nwh + 16384;                       // 32768      (swizzled)
  f16_t* swh    = mwh + 32768;                       // 32768      (swizzled)
  f16_t* awh    = swh + 32768;                       // 32768      (swizzled)

  auto swz = [&](const float* s, f16_t* d, int K, int J, int nmat) {
    int total = K * J * nmat;
    int blocks = (total + 255) / 256;
    if (blocks > 4096) blocks = 4096;
    swz_b16<<<blocks, 256, 0, stream>>>(s, d, K, J, nmat);
  };
  // cores: treat as 512 independent (K=128 x J=64) matrices (one per [c][i])
  swz(cores,  coresH, 128, 64, 512);
  swz(Amat,   AH,     128, 64, 64);
  swz(e2w,    e2h,    128, 128, 1);
  swz(nadew,  nwh,    64,  256, 1);
  swz(muw,    mwh,    256, 128, 1);
  swz(sigw,   swh,    256, 128, 1);
  swz(alphaw, awh,    256, 128, 1);
  cvt_f32_f16<<<4, 256, 0, stream>>>(e1w, e1h, 1024);
  init_state<<<256, 256, 0, stream>>>(initw, hws, out, N);

  StepParams P;
  P.X = X; P.h = hws;
  P.e1 = e1h; P.e2 = e2h; P.nw = nwh; P.mw = mwh; P.sw = swh; P.aw = awh;
  P.b1 = b1; P.b2 = b2; P.nb = nb; P.mb = mb; P.sb = sb; P.ab = ab;
  P.stats = stats; P.gamma = bng; P.beta = bnb;
  P.out = out;

  const size_t SMEM = 123904;   // ~121 KB dynamic LDS (WGP has 320 KB)

  for (int i = 0; i < 9; ++i) {
    if (i == 4) {               // i == l: step with A, then BatchNorm, then phi
      P.core = AH; P.colEnc = 3; P.colPhi = 4;
      fused_step<1><<<nblocks, 128, SMEM, stream>>>(P);
      bn_partial<<<NPART, 256, 0, stream>>>(hws, part, N);
      bn_final<<<1, 64, 0, stream>>>(part, stats, N, NPART);
      fused_step<2><<<nblocks, 128, SMEM, stream>>>(P);
    } else {
      int ci = (i == 0) ? 7 : (i - 1);    // torch's cores[-1] at i==0
      P.core = coresH + (size_t)ci * 524288;
      P.colEnc = (i == 0) ? 8 : (i - 1);
      P.colPhi = i;
      fused_step<0><<<nblocks, 128, SMEM, stream>>>(P);
    }
  }
}
